// SparseParallelLayer_51419348467812
// MI455X (gfx1250) — matrix operand order repaired and checked by reading, not compiled, not run on hardware
//
#include <hip/hip_runtime.h>
#include <hip/hip_bf16.h>

// ---------------- problem constants (from reference) ----------------
#define BSZ   2
#define SEQ   8192
#define NH    32
#define HD    128
#define DM    4096      // NH*HD
#define DQ    1024
#define QL    64
#define TGT   2048
#define NKV   32        // TGT/QL
#define NSINK 4
#define DQH   32        // DQ/NH

typedef __attribute__((ext_vector_type(16))) __bf16 v16bf;
typedef __attribute__((ext_vector_type(8)))  float  v8f;

union Frag { v16bf v; uint4 u[2]; };

// gfx1250 async copy: global -> LDS, 16 bytes per lane, tracked by ASYNCcnt
__device__ __forceinline__ void async_copy_b128(unsigned lds_off, const void* gaddr) {
  asm volatile("global_load_async_to_lds_b128 %0, %1, off"
               :: "v"(lds_off), "v"((unsigned long long)(size_t)gaddr)
               : "memory");
}
__device__ __forceinline__ void wait_async_le3() {
  asm volatile("s_wait_asynccnt 0x3" ::: "memory");
}
__device__ __forceinline__ void wait_async_0() {
  asm volatile("s_wait_asynccnt 0x0" ::: "memory");
}

// ---------------- kernel 0: fp32 -> bf16 streaming convert ----------------
__global__ __launch_bounds__(256) void cvt_f32_bf16(const float* __restrict__ src,
                                                    __bf16* __restrict__ dst, int n4) {
  int i = blockIdx.x * blockDim.x + threadIdx.x;
  if (i >= n4) return;
  const float4 x = reinterpret_cast<const float4*>(src)[i];
  size_t o = (size_t)i * 4;
  dst[o + 0] = (__bf16)x.x;
  dst[o + 1] = (__bf16)x.y;
  dst[o + 2] = (__bf16)x.z;
  dst[o + 3] = (__bf16)x.w;
}

// ---------------- kernel 1a: small direct-load GEMM (used for pq) ----------------
// C[M,N](bf16) = A[M,K](bf16) x B[N,K](bf16)^T ; one wave per 16x16 tile.
__global__ __launch_bounds__(256) void gemm_nt_bf16(const __bf16* __restrict__ A,
                                                    const __bf16* __restrict__ B,
                                                    __bf16* __restrict__ C,
                                                    int M, int N, int K) {
  int wave  = (blockIdx.x * blockDim.x + threadIdx.x) >> 5;
  int lane  = threadIdx.x & 31;
  int tilesN = N >> 4;
  int tm = wave / tilesN;
  int tn = wave - tm * tilesN;
  if (tm * 16 >= M) return;
  const int half = lane >> 4, l15 = lane & 15;

  const __bf16* arow = A + (size_t)(tm * 16 + l15) * K;
  const __bf16* brow = B + (size_t)(tn * 16 + l15) * K;

  v8f acc = {};
  for (int kk = 0; kk < K; kk += 32) {
    __builtin_prefetch(arow + kk + 128, 0, 1);
    __builtin_prefetch(brow + kk + 128, 0, 1);
    Frag af, bf;
    af.u[0] = *reinterpret_cast<const uint4*>(arow + kk + half * 8);
    af.u[1] = *reinterpret_cast<const uint4*>(arow + kk + 16 + half * 8);
    bf.u[0] = *reinterpret_cast<const uint4*>(brow + kk + half * 8);
    bf.u[1] = *reinterpret_cast<const uint4*>(brow + kk + 16 + half * 8);
    acc = __builtin_amdgcn_wmma_f32_16x16x32_bf16(false, af.v, false, bf.v,
                                                  (short)0, acc, false, false);
  }
  for (int r = 0; r < 8; ++r) {
    int m = tm * 16 + r + half * 8;
    C[(size_t)m * N + tn * 16 + l15] = (__bf16)acc[r];
  }
}

// ---------------- kernel 1b: LDS-staged blocked GEMM (used for pk) ----------------
// Block = 8 waves computes a 128x64 C tile. A/B K-slabs (BK=32) are staged into
// LDS via global_load_async_to_lds_b128, double-buffered on ASYNCcnt.
// Each wave: 16x64 strip = 4 accumulators; A-fragment reused across 4 WMMAs.
#define BM 128
#define BN 64
#define BK 32
__global__ __launch_bounds__(256) void gemm_nt_bf16_lds(const __bf16* __restrict__ A,
                                                        const __bf16* __restrict__ B,
                                                        __bf16* __restrict__ C,
                                                        int M, int N, int K) {
  __shared__ __bf16 sA[2][BM * BK];   // 8 KB per buffer
  __shared__ __bf16 sB[2][BN * BK];   // 4 KB per buffer

  const int tid  = threadIdx.x;
  const int bn_t = N / BN;
  const int bm   = blockIdx.x / bn_t;
  const int bn   = blockIdx.x - bm * bn_t;

  // --- per-thread async-copy assignments (16B each) ---
  const int rA0 = tid >> 2, cA = (tid & 3) * 8;            // A rows 0..63
  const int rA1 = rA0 + 64;                                 // A rows 64..127
  const int rB  = tid >> 2, cB = cA;                        // B rows 0..63
  const __bf16* gA0 = A + (size_t)(bm * BM + rA0) * K + cA;
  const __bf16* gA1 = A + (size_t)(bm * BM + rA1) * K + cA;
  const __bf16* gB  = B + (size_t)(bn * BN + rB) * K + cB;
  const unsigned lA0 = (unsigned)(size_t)&sA[0][rA0 * BK + cA];
  const unsigned lA1 = (unsigned)(size_t)&sA[0][rA1 * BK + cA];
  const unsigned lB  = (unsigned)(size_t)&sB[0][rB * BK + cB];

  // --- per-wave compute addressing ---
  const int wv = tid >> 5, lane = tid & 31;
  const int half = lane >> 4, l15 = lane & 15;
  const __bf16* aL = &sA[0][(wv * 16 + l15) * BK + half * 8];

  v8f acc0 = {}, acc1 = {}, acc2 = {}, acc3 = {};
  const int NS = K / BK;

  // prologue: stage 0
  async_copy_b128(lA0, gA0);
  async_copy_b128(lA1, gA1);
  async_copy_b128(lB,  gB);

  for (int s = 0; s < NS; ++s) {
    const unsigned cur = (s & 1) ? (unsigned)(BM * BK * 2) : 0u;   // byte offset of buffer s
    if (s + 1 < NS) {
      const unsigned nxt = ((s + 1) & 1) ? (unsigned)(BM * BK * 2) : 0u;
      const int kk = (s + 1) * BK;
      async_copy_b128(lA0 + nxt, gA0 + kk);
      async_copy_b128(lA1 + nxt, gA1 + kk);
      async_copy_b128(lB + (((s + 1) & 1) ? (unsigned)(BN * BK * 2) : 0u), gB + kk);
      wait_async_le3();                  // previous stage complete (in-order)
    } else {
      wait_async_0();
    }
    __syncthreads();                     // LDS tiles visible to all waves

    Frag af;
    const __bf16* aP = aL + (cur / 2);   // cur is bytes; aL is __bf16*
    af.u[0] = *reinterpret_cast<const uint4*>(aP);
    af.u[1] = *reinterpret_cast<const uint4*>(aP + 16);

    const unsigned curB = (s & 1) ? (unsigned)(BN * BK) : 0u;      // in elements
#pragma unroll
    for (int nsub = 0; nsub < 4; ++nsub) {
      Frag bf;
      const __bf16* bP = &sB[0][curB + (nsub * 16 + l15) * BK + half * 8];
      bf.u[0] = *reinterpret_cast<const uint4*>(bP);
      bf.u[1] = *reinterpret_cast<const uint4*>(bP + 16);
      if (nsub == 0) acc0 = __builtin_amdgcn_wmma_f32_16x16x32_bf16(false, af.v, false, bf.v, (short)0, acc0, false, false);
      if (nsub == 1) acc1 = __builtin_amdgcn_wmma_f32_16x16x32_bf16(false, af.v, false, bf.v, (short)0, acc1, false, false);
      if (nsub == 2) acc2 = __builtin_amdgcn_wmma_f32_16x16x32_bf16(false, af.v, false, bf.v, (short)0, acc2, false, false);
      if (nsub == 3) acc3 = __builtin_amdgcn_wmma_f32_16x16x32_bf16(false, af.v, false, bf.v, (short)0, acc3, false, false);
    }
    __syncthreads();                     // done reading before buffer reuse
  }

  // epilogue: D f32 layout -> bf16 store (vgpr r: M = r + 8*half, N = l15)
  const int mbase = bm * BM + wv * 16;
#pragma unroll
  for (int r = 0; r < 8; ++r) {
    const size_t m = (size_t)(mbase + r + half * 8);
    C[m * N + bn * BN + 0 * 16 + l15] = (__bf16)acc0[r];
    C[m * N + bn * BN + 1 * 16 + l15] = (__bf16)acc1[r];
    C[m * N + bn * BN + 2 * 16 + l15] = (__bf16)acc2[r];
    C[m * N + bn * BN + 3 * 16 + l15] = (__bf16)acc3[r];
  }
}

// ---------------- kernel 2: logits[b,h,q,k] = (1/32) * q_heads . k_heads ----------------
__global__ __launch_bounds__(256) void logits_wmma(const __bf16* __restrict__ pq,
                                                   const __bf16* __restrict__ pk,
                                                   float* __restrict__ logits) {
  int wave = (blockIdx.x * blockDim.x + threadIdx.x) >> 5;
  int lane = threadIdx.x & 31;
  const int KT = SEQ / 16;
  int kt = wave % KT;
  int rest = wave / KT;
  int qt = rest & 3;  rest >>= 2;
  int h  = rest & 31;
  int b  = rest >> 5;
  if (b >= BSZ) return;
  const int half = lane >> 4, l15 = lane & 15;

  const __bf16* arow = pq + (size_t)(qt * 16 + l15) * DQ + h * DQH;
  const __bf16* brow = pk + ((size_t)b * SEQ + kt * 16 + l15) * DQ + h * DQH;

  Frag af, bf;
  af.u[0] = *reinterpret_cast<const uint4*>(arow + half * 8);
  af.u[1] = *reinterpret_cast<const uint4*>(arow + 16 + half * 8);
  bf.u[0] = *reinterpret_cast<const uint4*>(brow + half * 8);
  bf.u[1] = *reinterpret_cast<const uint4*>(brow + 16 + half * 8);

  v8f acc = {};
  acc = __builtin_amdgcn_wmma_f32_16x16x32_bf16(false, af.v, false, bf.v,
                                                (short)0, acc, false, false);

  float* orow = logits + ((size_t)(b * NH + h) * QL) * SEQ;
  for (int r = 0; r < 8; ++r) {
    int q = qt * 16 + r + half * 8;
    orow[(size_t)q * SEQ + kt * 16 + l15] = acc[r] * 0.03125f;   // 1/sqrt(1024)
  }
}

// ---------------- kernel 3: per-row softmax + top-32 (one block per (b,h,q)) ----------------
__global__ __launch_bounds__(256) void softmax_topk(const float* __restrict__ logits,
                                                    float* __restrict__ tv,
                                                    int* __restrict__ ti) {
  __shared__ float sval[256];
  __shared__ int   sidx[256];
  const int t = threadIdx.x;
  const float* row = logits + (size_t)blockIdx.x * SEQ;

  float lv[SEQ / 256];
  float mx = -__builtin_inff();
  for (int i = 0; i < SEQ / 256; ++i) { float x = row[t + i * 256]; lv[i] = x; mx = fmaxf(mx, x); }
  sval[t] = mx; __syncthreads();
  for (int s = 128; s > 0; s >>= 1) { if (t < s) sval[t] = fmaxf(sval[t], sval[t + s]); __syncthreads(); }
  mx = sval[0]; __syncthreads();

  float sum = 0.f;
  for (int i = 0; i < SEQ / 256; ++i) { lv[i] = __expf(lv[i] - mx); sum += lv[i]; }
  sval[t] = sum; __syncthreads();
  for (int s = 128; s > 0; s >>= 1) { if (t < s) sval[t] += sval[t + s]; __syncthreads(); }
  const float inv = 1.f / sval[0]; __syncthreads();

  const int bh = blockIdx.x >> 6;
  const int q  = blockIdx.x & 63;
  float* ov = tv + (size_t)bh * TGT + q * NKV;
  int*   oi = ti + (size_t)bh * TGT + q * NKV;

  for (int s = 0; s < NKV; ++s) {
    float bm = -__builtin_inff(); int bi = SEQ;
    for (int i = 0; i < SEQ / 256; ++i) {
      int idx = t + i * 256;
      if (lv[i] > bm) { bm = lv[i]; bi = idx; }
    }
    sval[t] = bm; sidx[t] = bi; __syncthreads();
    for (int st = 128; st > 0; st >>= 1) {
      if (t < st) {
        bool take = (sval[t + st] > sval[t]) ||
                    (sval[t + st] == sval[t] && sidx[t + st] < sidx[t]);
        if (take) { sval[t] = sval[t + st]; sidx[t] = sidx[t + st]; }
      }
      __syncthreads();
    }
    int win = sidx[0];
    if (t == 0) { ov[s] = sval[0] * inv; oi[s] = win; }
    if ((win & 255) == t) lv[win >> 8] = -__builtin_inff();
    __syncthreads();
  }
}

// ---------------- kernel 4: sink replacement (one block per (b,h)) ----------------
__global__ __launch_bounds__(256) void sink_fix(float* __restrict__ tv, int* __restrict__ ti) {
  __shared__ float sval[256];
  __shared__ int   sidx[256];
  const int t = threadIdx.x;
  float* v  = tv + (size_t)blockIdx.x * TGT;
  int*   ix = ti + (size_t)blockIdx.x * TGT;

  float lv[TGT / 256];
  for (int i = 0; i < TGT / 256; ++i) lv[i] = v[t + i * 256];

  for (int s = 0; s < NSINK; ++s) {
    float bm = -__builtin_inff(); int bi = TGT;
    for (int i = 0; i < TGT / 256; ++i) {
      int idx = t + i * 256;
      if (lv[i] > bm) { bm = lv[i]; bi = idx; }
    }
    sval[t] = bm; sidx[t] = bi; __syncthreads();
    for (int st = 128; st > 0; st >>= 1) {
      if (t < st) {
        bool take = (sval[t + st] > sval[t]) ||
                    (sval[t + st] == sval[t] && sidx[t + st] < sidx[t]);
        if (take) { sval[t] = sval[t + st]; sidx[t] = sidx[t + st]; }
      }
      __syncthreads();
    }
    int win = sidx[0];
    if (t == 0) { ix[win] = s; v[win] = 1.0f; }
    if ((win & 255) == t) lv[win >> 8] = -__builtin_inff();
    __syncthreads();
  }
}

// ---------------- kernel 5: gather K/V (one wave per (b,t,h)) ----------------
__global__ __launch_bounds__(256) void gather_kv(const float* __restrict__ key,
                                                 const float* __restrict__ val,
                                                 const int* __restrict__ ti,
                                                 const float* __restrict__ tv,
                                                 float* __restrict__ out) {
  int wave = (blockIdx.x * blockDim.x + threadIdx.x) >> 5;
  int lane = threadIdx.x & 31;
  int h  = wave & 31;
  int rr = wave >> 5;
  int tt = rr & (TGT - 1);
  int b  = rr >> 11;
  if (b >= BSZ) return;

  size_t sel = (size_t)(b * NH + h) * TGT + tt;
  int   src  = ti[sel];
  float s    = tv[sel];
  float scale = (1.0f + s) - s;

  const size_t OUTK = (size_t)BSZ * TGT * NH * HD;
  const float4* ks = reinterpret_cast<const float4*>(key + (((size_t)b * SEQ + src) * NH + h) * HD);
  const float4* vs = reinterpret_cast<const float4*>(val + (((size_t)b * SEQ + src) * NH + h) * HD);
  float4* kd = reinterpret_cast<float4*>(out + (((size_t)b * TGT + tt) * NH + h) * HD);
  float4* vd = reinterpret_cast<float4*>(out + OUTK + (((size_t)b * TGT + tt) * NH + h) * HD);

  float4 kx = ks[lane];
  float4 vx = vs[lane];
  kx.x *= scale; kx.y *= scale; kx.z *= scale; kx.w *= scale;
  vx.x *= scale; vx.y *= scale; vx.z *= scale; vx.w *= scale;
  kd[lane] = kx;
  vd[lane] = vx;
}

// ---------------- host-side launch ----------------
extern "C" void kernel_launch(void* const* d_in, const int* in_sizes, int n_in,
                              void* d_out, int out_size, void* d_ws, size_t ws_size,
                              hipStream_t stream) {
  const float* key   = (const float*)d_in[0];
  const float* value = (const float*)d_in[1];
  const float* query = (const float*)d_in[2];
  const float* Wq    = (const float*)d_in[3];
  const float* Wk    = (const float*)d_in[4];
  float* out = (float*)d_out;

  char* ws = (char*)d_ws;
  constexpr size_t N_KEY = (size_t)BSZ * SEQ * DM;
  constexpr size_t N_WK  = (size_t)DQ * DM;
  constexpr size_t N_Q   = (size_t)QL * DQ;
  constexpr size_t N_WQ  = (size_t)DQ * DQ;
  constexpr size_t N_PK  = (size_t)BSZ * SEQ * DQ;
  constexpr size_t N_LOG = (size_t)BSZ * NH * QL * SEQ;
  constexpr size_t N_TK  = (size_t)BSZ * NH * TGT;

  size_t off = 0;
  auto alloc = [&](size_t bytes) { size_t o = off; off = (off + bytes + 255) & ~(size_t)255; return o; };
  __bf16* keyb = (__bf16*)(ws + alloc(N_KEY * 2));
  __bf16* wkb  = (__bf16*)(ws + alloc(N_WK * 2));
  __bf16* qb   = (__bf16*)(ws + alloc(N_Q * 2));
  __bf16* wqb  = (__bf16*)(ws + alloc(N_WQ * 2));
  __bf16* pqb  = (__bf16*)(ws + alloc(N_Q * 2));
  __bf16* pkb  = (__bf16*)(ws + alloc(N_PK * 2));
  float*  lgt  = (float*)(ws + alloc(N_LOG * 4));
  float*  tv   = (float*)(ws + alloc(N_TK * 4));
  int*    ti   = (int*)(ws + alloc(N_TK * 4));

  cvt_f32_bf16<<<(int)(N_KEY / 4 / 256), 256, 0, stream>>>(key,   keyb, (int)(N_KEY / 4));
  cvt_f32_bf16<<<(int)(N_WK  / 4 / 256), 256, 0, stream>>>(Wk,    wkb,  (int)(N_WK  / 4));
  cvt_f32_bf16<<<(int)(N_Q   / 4 / 256), 256, 0, stream>>>(query, qb,   (int)(N_Q   / 4));
  cvt_f32_bf16<<<(int)(N_WQ  / 4 / 256), 256, 0, stream>>>(Wq,    wqb,  (int)(N_WQ  / 4));

  // pq = query @ Wq^T : M=64, N=1024, K=1024 (direct-load path)
  gemm_nt_bf16<<<(QL / 16) * (DQ / 16) * 32 / 256, 256, 0, stream>>>(qb, wqb, pqb, QL, DQ, DQ);

  // pk = key_flat @ Wk^T : M=16384, N=1024, K=4096 (async-LDS blocked path)
  gemm_nt_bf16_lds<<<(BSZ * SEQ / BM) * (DQ / BN), 256, 0, stream>>>(keyb, wkb, pkb,
                                                                     BSZ * SEQ, DQ, DM);

  // logits
  logits_wmma<<<BSZ * NH * (QL / 16) * (SEQ / 16) * 32 / 256, 256, 0, stream>>>(pqb, pkb, lgt);

  // softmax + top-32
  softmax_topk<<<BSZ * NH * QL, 256, 0, stream>>>(lgt, tv, ti);

  // sink replacement
  sink_fix<<<BSZ * NH, 256, 0, stream>>>(tv, ti);

  // gather
  gather_kv<<<BSZ * TGT * NH * 32 / 256, 256, 0, stream>>>(key, value, ti, tv, out);
}
